// RES_HMM_33801392619809
// MI455X (gfx1250) — matrix operand order repaired and checked by reading, not compiled, not run on hardware
//
#include <hip/hip_runtime.h>

// ---------------------------------------------------------------------------
// Model dims
// ---------------------------------------------------------------------------
#define BB    64
#define T_IN  6000
#define TT    1000
#define DDIM  512
#define HHID  256
#define SSTS  5
#define KC3   1216      // 19*64, conv3 im2col reduction (19 * 64-K steps)

// ---------------------------------------------------------------------------
// bf16 helpers (manual RNE) + WMMA fragment types
// ---------------------------------------------------------------------------
__device__ __forceinline__ unsigned short f2bf(float f) {
  union { float f; unsigned u; } v; v.f = f;
  unsigned r = v.u + 0x7FFFu + ((v.u >> 16) & 1u);
  return (unsigned short)(r >> 16);
}
__device__ __forceinline__ float bf2f(unsigned short h) {
  union { unsigned u; float f; } v; v.u = ((unsigned)h) << 16;
  return v.f;
}
__device__ __forceinline__ float sigf(float x) { return 1.f / (1.f + __expf(-x)); }

typedef __attribute__((ext_vector_type(16))) __bf16 v16bf;
typedef __attribute__((ext_vector_type(8)))  float  v8f;

union Frag16 {
  v16bf v;
  unsigned short u[16];
  uint4 q[2];
};

// Invert 16-bit A-fragment K layout: kk(0..31) -> (half, j)
__device__ __forceinline__ void a_kk_to_hj(int kk, int& half, int& j) {
  if (kk < 8)       { half = 0; j = kk;      }
  else if (kk < 16) { half = 1; j = kk - 8;  }
  else if (kk < 24) { half = 0; j = kk - 8;  }
  else              { half = 1; j = kk - 16; }
}

// ---------------------------------------------------------------------------
// Tensor Data Mover: async 2-D tile load (global -> LDS), D# per ISA 08 §8.
// data_size=2B; tensor = [num_rows x row_len] bf16, row stride = row_len;
// tile = [tile_h x tile_w] starting at gptr; rows land contiguously in LDS.
// ---------------------------------------------------------------------------
#if defined(__gfx1250__)
typedef __attribute__((ext_vector_type(4))) unsigned int u32x4;
typedef __attribute__((ext_vector_type(8))) int i32x8;
typedef __attribute__((ext_vector_type(4))) int i32x4;

__device__ __forceinline__ void tdm_load_2d_to_lds(const void* gptr, unsigned lds_off,
                                                   int row_len, int num_rows,
                                                   int tile_w, int tile_h) {
  unsigned long long ga = (unsigned long long)gptr;
  u32x4 g0;
  g0[0] = 1u;                                            // count=1, user mode
  g0[1] = lds_off;                                       // LDS byte address
  g0[2] = (unsigned)(ga & 0xFFFFFFFFu);                  // global_addr[31:0]
  g0[3] = (unsigned)((ga >> 32) & 0x1FFFFFFu) | (2u << 30); // addr[56:32] | type=2
  i32x8 g1;
  g1[0] = 0x00010000;                                    // data_size = 2 bytes
  g1[1] = (int)(((unsigned)row_len & 0xFFFFu) << 16);    // tensor_dim0[15:0]
  g1[2] = (int)(((unsigned)row_len >> 16) |
                (((unsigned)num_rows & 0xFFFFu) << 16)); // dim0[31:16] | dim1[15:0]
  g1[3] = (int)(((unsigned)num_rows >> 16) |
                ((unsigned)tile_w << 16));               // dim1[31:16] | tile_dim0
  g1[4] = (int)((unsigned)tile_h & 0xFFFFu);             // tile_dim1 (tile_dim2=0)
  g1[5] = row_len;                                       // tensor_dim0_stride[31:0]
  g1[6] = 0;                                             // stride0[47:32] | stride1 lo
  g1[7] = 0;
  i32x4 gz = {0, 0, 0, 0};
#if defined(__clang_major__) && __clang_major__ >= 23
  i32x8 gz8 = {0, 0, 0, 0, 0, 0, 0, 0};
  __builtin_amdgcn_tensor_load_to_lds(g0, g1, gz, gz, gz8, 0);
#else
  __builtin_amdgcn_tensor_load_to_lds(g0, g1, gz, gz, 0);
#endif
}
#endif

// ---------------------------------------------------------------------------
// Pack f32 [K,N] weight into WMMA B-fragment-major bf16:
// dst[((kc*(N/16)+nt)*32 + lane)*16 + j],  kc=k>>5, half=(k>>3)&1, j=(k&7)+((k>>4)&1)*8,
// nt=n>>4, lane=half*16+(n&15).  (B-frag element K = j<8 ? half*8+j : 16+half*8+(j-8))
// ---------------------------------------------------------------------------
__global__ void k_pack_bfrag(const float* __restrict__ src, unsigned short* __restrict__ dst,
                             int K, int N) {
  int idx = blockIdx.x * blockDim.x + threadIdx.x;
  if (idx >= K * N) return;
  int k = idx / N, n = idx % N;
  int kc = k >> 5;
  int half = (k >> 3) & 1, j = (k & 7) + (((k >> 4) & 1) << 3);
  int nt = n >> 4, lane = half * 16 + (n & 15);
  dst[(((size_t)(kc * (N >> 4) + nt) * 32 + lane) << 4) + j] = f2bf(src[(size_t)k * N + n]);
}

// Fold batchnorm (g,b,mu,var) -> scale/shift. [0:64)=bn1, [64:128)=bn2, [128:512)=bn3
__global__ void k_bnfold(const float* g1, const float* b1, const float* m1, const float* v1,
                         const float* g2, const float* b2, const float* m2, const float* v2,
                         const float* g3, const float* b3, const float* m3, const float* v3,
                         float* scale, float* shift) {
  int i = blockIdx.x * blockDim.x + threadIdx.x;
  if (i < 64)       { float s = g1[i] * rsqrtf(v1[i] + 1e-5f); scale[i] = s; shift[i] = b1[i] - m1[i] * s; }
  else if (i < 128) { int j = i - 64;  float s = g2[j] * rsqrtf(v2[j] + 1e-5f); scale[i] = s; shift[i] = b2[j] - m2[j] * s; }
  else if (i < 512) { int j = i - 128; float s = g3[j] * rsqrtf(v3[j] + 1e-5f); scale[i] = s; shift[i] = b3[j] - m3[j] * s; }
}

// ---------------------------------------------------------------------------
// Conv1 (1->64, K=5, SAME) + swish + bn1  -> bf16
// ---------------------------------------------------------------------------
__global__ void k_conv1(const float* __restrict__ in, const float* __restrict__ w,
                        const float* __restrict__ bias, const float* __restrict__ bnS,
                        const float* __restrict__ bnB, unsigned short* __restrict__ out) {
  int idx = blockIdx.x * blockDim.x + threadIdx.x;
  if (idx >= BB * T_IN * 64) return;
  int o = idx & 63;
  int t = (idx >> 6) % T_IN;
  int b = idx / (T_IN * 64);
  float acc = bias[o];
#pragma unroll
  for (int k = 0; k < 5; ++k) {
    int tt = t + k - 2;
    if (tt >= 0 && tt < T_IN) acc += in[b * T_IN + tt] * w[k * 64 + o];
  }
  float sw = acc * sigf(acc);
  out[idx] = f2bf(sw * bnS[o] + bnB[o]);
}

// Conv2 (64->64, K=5, SAME) + swish + bn2 -> bf16
__global__ void k_conv2(const unsigned short* __restrict__ a1, const float* __restrict__ w,
                        const float* __restrict__ bias, const float* __restrict__ bnS,
                        const float* __restrict__ bnB, unsigned short* __restrict__ out) {
  int idx = blockIdx.x * blockDim.x + threadIdx.x;
  if (idx >= BB * T_IN * 64) return;
  int o = idx & 63;
  int t = (idx >> 6) % T_IN;
  int b = idx / (T_IN * 64);
  float acc = bias[o];
  for (int k = 0; k < 5; ++k) {
    int tt = t + k - 2;
    if (tt < 0 || tt >= T_IN) continue;
    const unsigned short* src = a1 + ((size_t)(b * T_IN + tt) << 6);
    const float* wr = w + (k << 6) * 64;  // w[(k*64+c)*64+o]
#pragma unroll 8
    for (int c = 0; c < 64; ++c) acc += bf2f(src[c]) * wr[c * 64 + o];
  }
  float sw = acc * sigf(acc);
  out[idx] = f2bf(sw * bnS[o] + bnB[o]);
}

// im2col for conv3: A3[r=(b*1000+t), col=(k*64+c)] = a2[b, t*6-6+k, c] (zero pad)
__global__ void k_im2col3(const unsigned short* __restrict__ a2, unsigned short* __restrict__ out) {
  int idx = blockIdx.x * blockDim.x + threadIdx.x;
  if (idx >= BB * TT * KC3) return;
  int col = idx % KC3;
  int r   = idx / KC3;
  int k = col >> 6, c = col & 63;
  int b = r / TT, t = r % TT;
  int tt = t * 6 - 6 + k;
  unsigned short v = 0;
  if (tt >= 0 && tt < T_IN) v = a2[((size_t)(b * T_IN + tt) << 6) + c];
  out[(size_t)idx] = v;
}

// ---------------------------------------------------------------------------
// WMMA bf16 GEMM: C = epi(A[M,K] * B[K,N] + bias).
// A row-major bf16 (TDM-staged 64x64 tiles -> LDS); B pre-packed frag-major.
// Block tile 64x128, 8 waves, each wave 16x64. M%64==0, N%128==0, K%64==0.
// ACT: 0=none 1=tanh 2=relu.  OUTMODE: 0=f32 row-major, 1=bf16 row-major,
// 2=bf16 in LSTM-scan tile swizzle (N must be 1024, M = 64*TT).
// ---------------------------------------------------------------------------
template <int ACT, bool HAS_BIAS, bool HAS_AFF, int OUTMODE>
__global__ __launch_bounds__(256) void k_wmma_gemm(
    const unsigned short* __restrict__ A, const unsigned short* __restrict__ Bp,
    const float* __restrict__ bias, const float* __restrict__ scale,
    const float* __restrict__ shift, float* __restrict__ Cf,
    unsigned short* __restrict__ Cbf, int M, int N, int K) {
  __shared__ __align__(16) unsigned short Atile[64 * 64];   // 8 KB, row-major
  int nb = N >> 7;
  int m0 = (blockIdx.x / nb) << 6;
  int n0 = (blockIdx.x % nb) << 7;
  int tid = threadIdx.x;
  int lane = tid & 31;
  int w = tid >> 5;
  int mt = w >> 1;          // 0..3
  int cg = w & 1;           // 0..1 (4 n-tiles each)
  int ntile_stride = N >> 4;
  v8f acc[4] = {};
#if defined(__gfx1250__)
  unsigned ldsA = (unsigned)(unsigned long long)(void*)Atile;
#endif
  for (int k0 = 0; k0 < K; k0 += 64) {
#if defined(__gfx1250__)
    if (w == 0) {
      tdm_load_2d_to_lds(A + (size_t)m0 * K + k0, ldsA, K, M, 64, 64);
    }
    __builtin_amdgcn_s_wait_tensorcnt(0);
#else
    for (int i = tid; i < 64 * 64 / 8; i += 256) {
      int ml = i >> 3, kb = i & 7;
      ((uint4*)Atile)[i] = *(const uint4*)(A + (size_t)(m0 + ml) * K + k0 + kb * 8);
    }
#endif
    __syncthreads();
#pragma unroll
    for (int kc2 = 0; kc2 < 2; ++kc2) {
      // A fragment straight from the row-major LDS tile (two b128 per lane):
      // lane(m,half): j0..7 -> kk = kc2*32 + half*8 + j ; j8..15 -> +16
      Frag16 a;
      int m = lane & 15, halfl = lane >> 4;
      const unsigned short* arow = &Atile[(mt * 16 + m) * 64 + kc2 * 32 + halfl * 8];
      a.q[0] = *(const uint4*)(arow);
      a.q[1] = *(const uint4*)(arow + 16);
      int kcg = (k0 >> 5) + kc2;
      Frag16 bfr[4];
#pragma unroll
      for (int i = 0; i < 4; ++i) {
        int nt = (n0 >> 4) + cg * 4 + i;
        const uint4* bp = (const uint4*)(Bp + (((size_t)(kcg * ntile_stride + nt) * 32 + lane) << 4));
        bfr[i].q[0] = bp[0];
        bfr[i].q[1] = bp[1];
      }
#pragma unroll
      for (int i = 0; i < 4; ++i) {
#if defined(__gfx1250__)
        acc[i] = __builtin_amdgcn_wmma_f32_16x16x32_bf16(false, a.v, false, bfr[i].v,
                                                         (short)0, acc[i], false, false);
#else
        (void)a; (void)bfr;
#endif
      }
    }
    __syncthreads();
  }
  // Epilogue (C layout: lane&15 -> N, vgpr v + 8*(lane>>4) -> M)
  int nn = lane & 15;
  int halfl = lane >> 4;
#pragma unroll
  for (int i = 0; i < 4; ++i) {
    int col = n0 + (cg * 4 + i) * 16 + nn;
    float bv = HAS_BIAS ? bias[col] : 0.f;
    float sc = HAS_AFF ? scale[col] : 1.f;
    float sh = HAS_AFF ? shift[col] : 0.f;
#pragma unroll
    for (int v = 0; v < 8; ++v) {
      int row = m0 + mt * 16 + v + 8 * halfl;
      float z = acc[i][v] + bv;
      if (ACT == 1) z = tanhf(z);
      else if (ACT == 2) z = fmaxf(z, 0.f);
      z = z * sc + sh;
      if (OUTMODE == 0) {
        Cf[(size_t)row * N + col] = z;
      } else if (OUTMODE == 1) {
        Cbf[(size_t)row * N + col] = f2bf(z);
      } else {
        // LSTM-scan tile swizzle: row = b*TT + t, tile (mts,nts), elem (lanes,vs)
        int b = row / TT, t = row - b * TT;
        int mi = b & 15;
        int vs = mi & 7, hs = mi >> 3;
        int lanes = hs * 16 + (col & 15);
        int nts = col >> 4, mts = b >> 4;
        Cbf[(((size_t)t * 256 + mts * 64 + nts) * 32 + lanes) * 8 + vs] = f2bf(z);
      }
    }
  }
}

// ---------------------------------------------------------------------------
// Persistent BiLSTM scan: blockIdx.x = direction (0 fwd, 1 bwd), 1024 threads.
// Per step: z = xw_sw[t] + h@Wh via WMMA (32 waves x 8 tiles over [64x1024]),
// fused gates, h kept in LDS in A-fragment swizzle, c in registers.
// xw is pre-swizzled per step tile (one b128 per lane per tile).
// Dynamic LDS: zl bf16 [64][1024] (128KB) + hsw bf16 [64*256] (32KB).
// ---------------------------------------------------------------------------
__global__ __launch_bounds__(1024) void k_lstm_scan(
    const unsigned short* __restrict__ xw_f, const unsigned short* __restrict__ xw_b,
    const unsigned short* __restrict__ whp_f, const unsigned short* __restrict__ whp_b,
    unsigned short* __restrict__ y_out) {
  extern __shared__ char smem[];
  unsigned short* zl  = (unsigned short*)smem;             // 64*1024
  unsigned short* hsw = (unsigned short*)(smem + 131072);  // 64*256
  const int dir = blockIdx.x;
  const unsigned short* xw  = dir ? xw_b  : xw_f;
  const unsigned short* whp = dir ? whp_b : whp_f;
  const int chan = dir * 256;
  int tid = threadIdx.x;
  int lane = tid & 31;
  int w = tid >> 5;
  int mt = w & 3;                 // batch row-tile (64 rows / 4 tiles)
  int ntbase = (w >> 2) * 8;      // 8 n-tiles per wave (64 total)
  int nn = lane & 15;
  int half = lane >> 4;
  float c[16];
#pragma unroll
  for (int i = 0; i < 16; ++i) c[i] = 0.f;
  for (int i = tid; i < 64 * 256; i += 1024) hsw[i] = 0;
  __syncthreads();

  for (int s = 0; s < TT; ++s) {
    int t = dir ? (TT - 1 - s) : s;
    v8f acc[8] = {};
    if (s > 0) {
#pragma unroll 1
      for (int kc = 0; kc < 8; ++kc) {
        Frag16 a;
        const uint4* ap = (const uint4*)&hsw[((mt * 8 + kc) * 32 + lane) << 4];
        a.q[0] = ap[0]; a.q[1] = ap[1];
#pragma unroll
        for (int i = 0; i < 8; ++i) {
          int nt = ntbase + i;
          Frag16 bf;
          const uint4* bp = (const uint4*)(whp + (size_t)(((kc * 64 + nt) * 32 + lane) << 4));
          bf.q[0] = bp[0]; bf.q[1] = bp[1];
#if defined(__gfx1250__)
          acc[i] = __builtin_amdgcn_wmma_f32_16x16x32_bf16(false, a.v, false, bf.v,
                                                           (short)0, acc[i], false, false);
#else
          (void)a; (void)bf;
#endif
        }
      }
    }
    // z = acc + xw_sw[t, tile] -> LDS   (one b128 per lane per tile)
#pragma unroll
    for (int i = 0; i < 8; ++i) {
      int nt = ntbase + i;
      uint4 xq = *(const uint4*)(xw + (((size_t)t * 256 + mt * 64 + nt) * 32 + lane) * 8);
      const unsigned short* xs = (const unsigned short*)&xq;
      int col = nt * 16 + nn;
#pragma unroll
      for (int v = 0; v < 8; ++v) {
        int b = mt * 16 + v + 8 * half;
        float z = acc[i][v] + bf2f(xs[v]);
        zl[(b << 10) + col] = f2bf(z);
      }
    }
    __syncthreads();
    // Fused gates: each thread owns 16 (b,j) cells, c in registers
#pragma unroll 1
    for (int i = 0; i < 16; ++i) {
      int idx = tid + (i << 10);
      int b = idx >> 8;
      int j = idx & 255;
      float zi = bf2f(zl[(b << 10) + j]);
      float zf = bf2f(zl[(b << 10) + 256 + j]);
      float zg = bf2f(zl[(b << 10) + 512 + j]);
      float zo = bf2f(zl[(b << 10) + 768 + j]);
      float ci = sigf(zf) * c[i] + sigf(zi) * tanhf(zg);
      c[i] = ci;
      float h = sigf(zo) * tanhf(ci);
      unsigned short hb = f2bf(h);
      int kc = j >> 5, kk = j & 31;
      int hf, jj; a_kk_to_hj(kk, hf, jj);
      int ln = hf * 16 + (b & 15);
      hsw[((((b >> 4) * 8 + kc) * 32 + ln) << 4) + jj] = hb;
      y_out[(size_t)(b * TT + t) * 512 + chan + j] = hb;
    }
    __syncthreads();
  }
}

// ---------------------------------------------------------------------------
// Row LayerNorm (512) f32 -> bf16
// ---------------------------------------------------------------------------
__global__ __launch_bounds__(256) void k_layernorm_bf16(
    const float* __restrict__ X, const float* __restrict__ g,
    const float* __restrict__ bta, unsigned short* __restrict__ out) {
  __shared__ float red[256];
  int r = blockIdx.x, tid = threadIdx.x;
  float v0 = X[(size_t)r * 512 + tid];
  float v1 = X[(size_t)r * 512 + 256 + tid];
  red[tid] = v0 + v1; __syncthreads();
  for (int st = 128; st > 0; st >>= 1) { if (tid < st) red[tid] += red[tid + st]; __syncthreads(); }
  float mu = red[0] * (1.f / 512.f); __syncthreads();
  float d0 = v0 - mu, d1 = v1 - mu;
  red[tid] = d0 * d0 + d1 * d1; __syncthreads();
  for (int st = 128; st > 0; st >>= 1) { if (tid < st) red[tid] += red[tid + st]; __syncthreads(); }
  float rstd = rsqrtf(red[0] * (1.f / 512.f) + 1e-12f);
  out[(size_t)r * 512 + tid]       = f2bf(d0 * rstd * g[tid] + bta[tid]);
  out[(size_t)r * 512 + 256 + tid] = f2bf(d1 * rstd * g[tid + 256] + bta[tid + 256]);
}

// reduce: y_red[r,s] = y2[r,:] @ reduce_W[:,s]   (N=5) -- one thread per row,
// row read with b128 loads.
__global__ void k_reduce(const unsigned short* __restrict__ y2,
                         const float* __restrict__ Wr, float* __restrict__ out) {
  int r = blockIdx.x * blockDim.x + threadIdx.x;
  if (r >= BB * TT) return;
  float acc[SSTS] = {0.f, 0.f, 0.f, 0.f, 0.f};
  const uint4* row = (const uint4*)(y2 + (size_t)r * 512);
  for (int dq = 0; dq < 64; ++dq) {
    uint4 q = row[dq];
    const unsigned short* e = (const unsigned short*)&q;
#pragma unroll
    for (int k = 0; k < 8; ++k) {
      float v = bf2f(e[k]);
      int d = dq * 8 + k;
#pragma unroll
      for (int s = 0; s < SSTS; ++s) acc[s] += v * Wr[d * SSTS + s];
    }
  }
#pragma unroll
  for (int s = 0; s < SSTS; ++s) out[(size_t)r * SSTS + s] = acc[s];
}

// ---------------------------------------------------------------------------
// HMM forward-backward log posteriors (5-state). 64 threads = batch.
// ---------------------------------------------------------------------------
__global__ __launch_bounds__(64) void k_hmm(const float* __restrict__ logits,
                                            const float* __restrict__ trans,
                                            float* __restrict__ alpha,
                                            float* __restrict__ post) {
  int b = threadIdx.x;
  float logA[SSTS][SSTS];
  for (int i = 0; i < SSTS; ++i) {
    float row[SSTS]; float mx = -1e30f;
    for (int j = 0; j < SSTS; ++j) {
      int d = j - i;
      float v = (d >= 0 && d <= 2) ? trans[d] : -1e30f;
      row[j] = v; mx = fmaxf(mx, v);
    }
    float se = 0.f;
    for (int j = 0; j < SSTS; ++j) se += __expf(row[j] - mx);
    float l = __logf(se) + mx;
    for (int j = 0; j < SSTS; ++j) logA[i][j] = row[j] - l;
  }
  float a[SSTS];
  {
    const float* lr = logits + (size_t)(b * TT) * SSTS;
    float mx = lr[0]; for (int s = 1; s < SSTS; ++s) mx = fmaxf(mx, lr[s]);
    float se = 0.f;   for (int s = 0; s < SSTS; ++s) se += __expf(lr[s] - mx);
    float lz = __logf(se) + mx;
    for (int s = 0; s < SSTS; ++s) { a[s] = lr[s] - lz - 1.6094379124341003f; alpha[(0 * BB + b) * SSTS + s] = a[s]; }
  }
  for (int t = 1; t < TT; ++t) {
    const float* lr = logits + (size_t)(b * TT + t) * SSTS;
    float mx = lr[0]; for (int s = 1; s < SSTS; ++s) mx = fmaxf(mx, lr[s]);
    float se = 0.f;   for (int s = 0; s < SSTS; ++s) se += __expf(lr[s] - mx);
    float lz = __logf(se) + mx;
    float an[SSTS];
    for (int j = 0; j < SSTS; ++j) {
      float tmp[SSTS]; float m2 = -1e30f;
      for (int i = 0; i < SSTS; ++i) { tmp[i] = a[i] + logA[i][j]; m2 = fmaxf(m2, tmp[i]); }
      float s2 = 0.f; for (int i = 0; i < SSTS; ++i) s2 += __expf(tmp[i] - m2);
      an[j] = __logf(s2) + m2 + (lr[j] - lz);
    }
    for (int j = 0; j < SSTS; ++j) { a[j] = an[j]; alpha[(t * BB + b) * SSTS + j] = an[j]; }
  }
  float mz = a[0]; for (int s = 1; s < SSTS; ++s) mz = fmaxf(mz, a[s]);
  float sz = 0.f;  for (int s = 0; s < SSTS; ++s) sz += __expf(a[s] - mz);
  float logZ = __logf(sz) + mz;
  float bt[SSTS];
  for (int s = 0; s < SSTS; ++s) {
    bt[s] = 0.f;
    post[(size_t)(b * TT + TT - 1) * SSTS + s] = alpha[((TT - 1) * BB + b) * SSTS + s] - logZ;
  }
  for (int t = TT - 2; t >= 0; --t) {
    const float* lr = logits + (size_t)(b * TT + t + 1) * SSTS;
    float mx = lr[0]; for (int s = 1; s < SSTS; ++s) mx = fmaxf(mx, lr[s]);
    float se = 0.f;   for (int s = 0; s < SSTS; ++s) se += __expf(lr[s] - mx);
    float lz = __logf(se) + mx;
    float e1[SSTS]; for (int s = 0; s < SSTS; ++s) e1[s] = lr[s] - lz;
    float bn[SSTS];
    for (int i = 0; i < SSTS; ++i) {
      float tmp[SSTS]; float m2 = -1e30f;
      for (int j = 0; j < SSTS; ++j) { tmp[j] = logA[i][j] + e1[j] + bt[j]; m2 = fmaxf(m2, tmp[j]); }
      float s2 = 0.f; for (int j = 0; j < SSTS; ++j) s2 += __expf(tmp[j] - m2);
      bn[i] = __logf(s2) + m2;
    }
    for (int j = 0; j < SSTS; ++j) {
      bt[j] = bn[j];
      post[(size_t)(b * TT + t) * SSTS + j] = alpha[(t * BB + b) * SSTS + j] + bn[j] - logZ;
    }
  }
}

// ---------------------------------------------------------------------------
// Final fuse: x + (post@expand_W)*relu_mlp, LayerNorm, unembed (N=5)
// ---------------------------------------------------------------------------
__global__ __launch_bounds__(256) void k_final(
    const float* __restrict__ x, const float* __restrict__ post,
    const float* __restrict__ expandW, const float* __restrict__ mlp,
    const float* __restrict__ g, const float* __restrict__ bta,
    const float* __restrict__ unembedW, float* __restrict__ out) {
  __shared__ float xrow[512];
  __shared__ float red[256];
  int r = blockIdx.x, tid = threadIdx.x;
  float p0 = post[(size_t)r * 5 + 0], p1 = post[(size_t)r * 5 + 1],
        p2 = post[(size_t)r * 5 + 2], p3 = post[(size_t)r * 5 + 3],
        p4 = post[(size_t)r * 5 + 4];
  float v[2];
#pragma unroll
  for (int h = 0; h < 2; ++h) {
    int d = tid + h * 256;
    float ye = p0 * expandW[d] + p1 * expandW[512 + d] + p2 * expandW[1024 + d] +
               p3 * expandW[1536 + d] + p4 * expandW[2048 + d];
    v[h] = x[(size_t)r * 512 + d] + ye * mlp[(size_t)r * 512 + d];
  }
  red[tid] = v[0] + v[1]; __syncthreads();
  for (int st = 128; st > 0; st >>= 1) { if (tid < st) red[tid] += red[tid + st]; __syncthreads(); }
  float mu = red[0] * (1.f / 512.f); __syncthreads();
  float d0 = v[0] - mu, d1 = v[1] - mu;
  red[tid] = d0 * d0 + d1 * d1; __syncthreads();
  for (int st = 128; st > 0; st >>= 1) { if (tid < st) red[tid] += red[tid + st]; __syncthreads(); }
  float rstd = rsqrtf(red[0] * (1.f / 512.f) + 1e-12f); __syncthreads();
  xrow[tid]       = d0 * rstd * g[tid] + bta[tid];
  xrow[tid + 256] = d1 * rstd * g[tid + 256] + bta[tid + 256];
  __syncthreads();
  if (tid < SSTS) {
    float acc = 0.f;
    for (int d = 0; d < 512; ++d) acc += xrow[d] * unembedW[d * SSTS + tid];
    out[(size_t)r * SSTS + tid] = acc;
  }
}

// ---------------------------------------------------------------------------
// Host launcher
// ---------------------------------------------------------------------------
extern "C" void kernel_launch(void* const* d_in, const int* in_sizes, int n_in,
                              void* d_out, int out_size, void* d_ws, size_t ws_size,
                              hipStream_t stream) {
  (void)in_sizes; (void)out_size; (void)ws_size;
  // ---- Parameter table (42 leaves) ----
  const float* P[42];
  if (n_in >= 42) {
    for (int i = 0; i < 42; ++i) P[i] = (const float*)d_in[i];
  } else {
    static const int  leafCounts[20] = {1,1,1,4,1,1,4,1,1,4,1,2,12,1,1,1,1,1,2,1};
    static const long leafSizes[42] = {
      64L*6000, 5*64, 64, 64,64,64,64, 5L*64*64, 64, 64,64,64,64,
      19L*64*384, 384, 384,384,384,384, 384L*512, 512,512,
      512L*1024, 256L*1024, 1024, 512L*1024, 256L*1024, 1024,
      512L*1024, 256L*1024, 1024, 512L*1024, 256L*1024, 1024,
      512L*5, 3, 5L*512, 512L*512, 512, 512,512, 512L*5
    };
    int leaf = 0;
    for (int e = 0; e < 20 && e < n_in; ++e) {
      const float* base = (const float*)d_in[e];
      long off2 = 0;
      for (int c = 0; c < leafCounts[e] && leaf < 42; ++c) {
        P[leaf] = base + off2;
        off2 += leafSizes[leaf];
        ++leaf;
      }
    }
  }

  // ---- Workspace layout ----
  char* W = (char*)d_ws;
  size_t off = 0;
  auto alloc = [&](size_t bytes) { size_t o = off; off += (bytes + 255) & ~(size_t)255; return o; };

  unsigned short* wEmbed = (unsigned short*)(W + alloc(384L * 512 * 2));
  unsigned short* wConv3 = (unsigned short*)(W + alloc((size_t)KC3 * 384 * 2));
  unsigned short* wWx[4]; for (int i = 0; i < 4; ++i) wWx[i] = (unsigned short*)(W + alloc(512L * 1024 * 2));
  unsigned short* wWhP[4]; for (int i = 0; i < 4; ++i) wWhP[i] = (unsigned short*)(W + alloc(256L * 1024 * 2));
  unsigned short* wMlp  = (unsigned short*)(W + alloc(512L * 512 * 2));
  float* bnScale = (float*)(W + alloc(512 * 4));
  float* bnShift = (float*)(W + alloc(512 * 4));
  float* x_embed = (float*)(W + alloc(64000L * 512 * 4));
  unsigned short* y_ln = (unsigned short*)(W + alloc(64000L * 512 * 2));
  unsigned short* y1   = (unsigned short*)(W + alloc(64000L * 512 * 2));
  unsigned short* y2   = (unsigned short*)(W + alloc(64000L * 512 * 2));
  float* mlp_out = (float*)(W + alloc(64000L * 512 * 4));
  float* y_red   = (float*)(W + alloc(320000L * 4));
  float* postB   = (float*)(W + alloc(320000L * 4));
  float* alphaB  = (float*)(W + alloc(320000L * 4));
  // BIG region: conv scratch, later aliased by xw buffers (conv scratch dead by then)
  size_t bigOff = alloc(303200000);
  unsigned short* a1 = (unsigned short*)(W + bigOff);
  unsigned short* a2 = a1 + 64L * 6000 * 64;
  unsigned short* A3 = a2 + 64L * 6000 * 64;
  unsigned short* a3 = A3 + 64000L * KC3;
  unsigned short* xw_f = (unsigned short*)(W + bigOff);
  unsigned short* xw_b = xw_f + 64000L * 1024;

  // ---- Weight packing / bn folding ----
  k_bnfold<<<2, 256, 0, stream>>>(P[3], P[4], P[5], P[6], P[9], P[10], P[11], P[12],
                                  P[15], P[16], P[17], P[18], bnScale, bnShift);
  k_pack_bfrag<<<(384 * 512 + 255) / 256, 256, 0, stream>>>(P[19], wEmbed, 384, 512);
  k_pack_bfrag<<<(KC3 * 384 + 255) / 256, 256, 0, stream>>>(P[13], wConv3, KC3, 384);
  for (int i = 0; i < 4; ++i) {
    k_pack_bfrag<<<(512 * 1024 + 255) / 256, 256, 0, stream>>>(P[22 + 3 * i], wWx[i], 512, 1024);
    k_pack_bfrag<<<(256 * 1024 + 255) / 256, 256, 0, stream>>>(P[23 + 3 * i], wWhP[i], 256, 1024);
  }
  k_pack_bfrag<<<(512 * 512 + 255) / 256, 256, 0, stream>>>(P[37], wMlp, 512, 512);

  // ---- Conv stack ----
  const int convN = BB * T_IN * 64;
  k_conv1<<<(convN + 255) / 256, 256, 0, stream>>>(P[0], P[1], P[2], bnScale, bnShift, a1);
  k_conv2<<<(convN + 255) / 256, 256, 0, stream>>>(a1, P[7], P[8], bnScale + 64, bnShift + 64, a2);
  k_im2col3<<<(BB * TT * KC3 + 255) / 256, 256, 0, stream>>>(a2, A3);
  // conv3 as WMMA GEMM (TDM-staged A): tanh + bn3 fold -> bf16
  k_wmma_gemm<1, true, true, 1><<<dim3(1000 * 3), 256, 0, stream>>>(
      A3, wConv3, P[14], bnScale + 128, bnShift + 128, nullptr, a3, 64000, 384, KC3);
  // embed GEMM -> f32 residual input
  k_wmma_gemm<0, false, false, 0><<<dim3(1000 * 4), 256, 0, stream>>>(
      a3, wEmbed, nullptr, nullptr, nullptr, x_embed, nullptr, 64000, 512, 384);
  // ln0 -> bf16
  k_layernorm_bf16<<<64000, 256, 0, stream>>>(x_embed, P[20], P[21], y_ln);

  // ---- BiLSTM layer 0 (xw written pre-swizzled for the scan) ----
  k_wmma_gemm<0, true, false, 2><<<dim3(1000 * 8), 256, 0, stream>>>(
      y_ln, wWx[0], P[24], nullptr, nullptr, nullptr, xw_f, 64000, 1024, 512);
  k_wmma_gemm<0, true, false, 2><<<dim3(1000 * 8), 256, 0, stream>>>(
      y_ln, wWx[1], P[27], nullptr, nullptr, nullptr, xw_b, 64000, 1024, 512);
  k_lstm_scan<<<2, 1024, 163840, stream>>>(xw_f, xw_b, wWhP[0], wWhP[1], y1);
  // ---- BiLSTM layer 1 ----
  k_wmma_gemm<0, true, false, 2><<<dim3(1000 * 8), 256, 0, stream>>>(
      y1, wWx[2], P[30], nullptr, nullptr, nullptr, xw_f, 64000, 1024, 512);
  k_wmma_gemm<0, true, false, 2><<<dim3(1000 * 8), 256, 0, stream>>>(
      y1, wWx[3], P[33], nullptr, nullptr, nullptr, xw_b, 64000, 1024, 512);
  k_lstm_scan<<<2, 1024, 163840, stream>>>(xw_f, xw_b, wWhP[2], wWhP[3], y2);

  // ---- HMM head ----
  k_reduce<<<(BB * TT + 255) / 256, 256, 0, stream>>>(y2, P[34], y_red);
  k_hmm<<<1, 64, 0, stream>>>(y_red, P[35], alphaB, postB);
  // mlp GEMM (bias + relu) -> f32
  k_wmma_gemm<2, true, false, 0><<<dim3(1000 * 4), 256, 0, stream>>>(
      y2, wMlp, P[38], nullptr, nullptr, mlp_out, nullptr, 64000, 512, 512);
  // ---- Final fuse + LN + unembed ----
  k_final<<<64000, 256, 0, stream>>>(x_embed, postB, P[36], mlp_out, P[39], P[40], P[41],
                                     (float*)d_out);
}
